// MinkUNet_18588618457312
// MI455X (gfx1250) — matrix operand order repaired and checked by reading, not compiled, not run on hardware
//
#include <hip/hip_runtime.h>
#include <hip/hip_bf16.h>

typedef __attribute__((ext_vector_type(16))) __bf16 v16bf;
typedef __attribute__((ext_vector_type(8)))  float  v8f;

union Frag {
    uint4 u[2];
    v16bf v;
};

// ---------------------------------------------------------------------------
// zero a float buffer (BN statistics accumulators + the zero-pad row)
// ---------------------------------------------------------------------------
__global__ void zero_f32(float* __restrict__ p, int n) {
    int i = blockIdx.x * blockDim.x + threadIdx.x;
    if (i < n) p[i] = 0.0f;
}

// ---------------------------------------------------------------------------
// Pre-swizzle a [taps,32,32] f32 weight tensor into WMMA B-fragment order,
// converted to bf16.  Fragment (k, t, lane) holds 16 bf16 elements:
//   e in [0,16):  ci = (e&7) + (lane>=16 ? 8 : 0) + (e>=8 ? 16 : 0),  co = (lane&15) + 16*t
// so the conv kernel can read one contiguous 32-byte chunk per lane.
// ---------------------------------------------------------------------------
__global__ void prep_w(const float* __restrict__ W, __bf16* __restrict__ frag,
                       int taps) {
    int i = blockIdx.x * blockDim.x + threadIdx.x;
    int total = taps * 1024;             // taps * 2 tiles * 32 lanes * 16 elems
    if (i >= total) return;
    int e    = i & 15;
    int lane = (i >> 4) & 31;
    int t    = (i >> 9) & 1;
    int k    = i >> 10;
    int ci   = (e & 7) + ((lane >> 4) << 3) + ((e >> 3) << 4);
    int co   = (lane & 15) + (t << 4);
    frag[i] = (__bf16)W[(k * 32 + ci) * 32 + co];
}

// ---------------------------------------------------------------------------
// Stem1: CIN=4 -> C0=32, 27 taps, plain VALU (memory-shaped, ~6% of FLOPs).
// One thread per (point, out-channel).  Accumulates BN sums via LDS+atomics.
// ---------------------------------------------------------------------------
__global__ void stem1_conv(const float* __restrict__ vf,
                           const int*   __restrict__ nbr,
                           const float* __restrict__ W,   // [27,4,32]
                           float* __restrict__ raw,
                           float* __restrict__ stat,      // [sum32 | sq32]
                           int N) {
    __shared__ float ssum[32];
    __shared__ float ssq[32];
    int tid = threadIdx.x;
    if (tid < 32) { ssum[tid] = 0.0f; ssq[tid] = 0.0f; }
    __syncthreads();

    long long g = (long long)blockIdx.x * blockDim.x + tid;
    int co = (int)(g & 31);
    long long pt = g >> 5;
    if (pt < N) {
        float acc = 0.0f;
        for (int k = 0; k < 27; ++k) {
            int idx = nbr[(long long)k * N + pt];
            if (idx >= 0) {
                const float* f = vf + (long long)idx * 4;
                const float* w = W + k * 128 + co;
                acc = fmaf(f[0], w[0],  acc);
                acc = fmaf(f[1], w[32], acc);
                acc = fmaf(f[2], w[64], acc);
                acc = fmaf(f[3], w[96], acc);
            }
        }
        raw[g] = acc;
        atomicAdd(&ssum[co], acc);
        atomicAdd(&ssq[co],  acc * acc);
    }
    __syncthreads();
    if (tid < 32) {
        atomicAdd(&stat[tid],      ssum[tid]);
        atomicAdd(&stat[32 + tid], ssq[tid]);
    }
}

// ---------------------------------------------------------------------------
// A-fragment gather with zero-pad row: feature buffer has a zeroed 32-channel
// row at the front, rows addressed as (idx+1).  idx == -1 lands on the zero
// row; the +1 row shift folds into the instruction immediate offset, so the
// gather is select-free (no v_cmp / v_max / v_cndmask, no WMMA WAR hazards).
// ---------------------------------------------------------------------------
__device__ __forceinline__ void load_a(const char* __restrict__ featb, int idx,
                                       int halfoff, Frag& fa) {
    const char* fp = featb + (ptrdiff_t)(idx * 64 + halfoff);
    fa.u[0] = *(const uint4*)(fp + 64);      // pad-row shift in IOFFSET
    fa.u[1] = *(const uint4*)(fp + 96);
}

// ---------------------------------------------------------------------------
// WMMA sparse conv: C0=32 -> C0=32, `taps` gather offsets.
// One wave computes a 64-point x 32-channel tile (four 16-row M-tiles):
// 8 v_wmma_f32_16x16x32_bf16 per tap, each LDS B fragment reused by all four
// M-tiles, 8 independent b128 gather streams per tap for memory ILP.
// Rows past N compute garbage from a clamped neighbor entry; WMMA rows are
// independent, and those rows are neither stored nor counted in BN stats.
// ---------------------------------------------------------------------------
__global__ void wmma_conv(const __bf16* __restrict__ featPad, // incl. pad row
                          const int*    __restrict__ nbr,
                          const uint4*  __restrict__ wfrag,   // B frags
                          float* __restrict__ raw,
                          float* __restrict__ stat,           // [sum32 | sq32]
                          int N, int taps) {
    extern __shared__ uint4 lds[];
    const int tid = threadIdx.x;

    // cooperative stage of all weight fragments: taps*2*32 frags * 32B
    const int nU4 = taps * 128;
    for (int i = tid; i < nU4; i += blockDim.x) lds[i] = wfrag[i];
    __syncthreads();

    const int lane  = tid & 31;
    const int wave  = tid >> 5;
    const int base0 = (blockIdx.x * 8 + wave) * 64;   // 4 M-tiles of 16
    if (base0 >= N) return;               // wave-uniform; no barriers follow

    const int row  = lane & 15;
    const int half = lane >> 4;
    const int halfoff = half * 16;                    // byte offset of K-run
    const char* featb = (const char*)featPad;

    int pc[4];       // clamped row index (always a valid nbr entry)
#pragma unroll
    for (int t = 0; t < 4; ++t) {
        int p = base0 + t * 16 + row;
        pc[t] = (p < N) ? p : (N - 1);
    }

    v8f c0[4] = {v8f{}, v8f{}, v8f{}, v8f{}};   // co [0,16)
    v8f c1[4] = {v8f{}, v8f{}, v8f{}, v8f{}};   // co [16,32)

    const int* __restrict__ nrow = nbr;
    for (int k = 0; k < taps; ++k) {
        int idx0 = nrow[pc[0]];
        int idx1 = nrow[pc[1]];
        int idx2 = nrow[pc[2]];
        int idx3 = nrow[pc[3]];
        nrow += N;

        Frag fa0, fa1, fa2, fa3;
        load_a(featb, idx0, halfoff, fa0);
        load_a(featb, idx1, halfoff, fa1);
        load_a(featb, idx2, halfoff, fa2);
        load_a(featb, idx3, halfoff, fa3);

        Frag fb0, fb1;
        int kt0 = ((k * 2 + 0) * 32 + lane) * 2;
        int kt1 = ((k * 2 + 1) * 32 + lane) * 2;
        fb0.u[0] = lds[kt0];  fb0.u[1] = lds[kt0 + 1];
        fb1.u[0] = lds[kt1];  fb1.u[1] = lds[kt1 + 1];

        c0[0] = __builtin_amdgcn_wmma_f32_16x16x32_bf16(false, fa0.v, false, fb0.v,
                                                        (short)0, c0[0], false, false);
        c1[0] = __builtin_amdgcn_wmma_f32_16x16x32_bf16(false, fa0.v, false, fb1.v,
                                                        (short)0, c1[0], false, false);
        c0[1] = __builtin_amdgcn_wmma_f32_16x16x32_bf16(false, fa1.v, false, fb0.v,
                                                        (short)0, c0[1], false, false);
        c1[1] = __builtin_amdgcn_wmma_f32_16x16x32_bf16(false, fa1.v, false, fb1.v,
                                                        (short)0, c1[1], false, false);
        c0[2] = __builtin_amdgcn_wmma_f32_16x16x32_bf16(false, fa2.v, false, fb0.v,
                                                        (short)0, c0[2], false, false);
        c1[2] = __builtin_amdgcn_wmma_f32_16x16x32_bf16(false, fa2.v, false, fb1.v,
                                                        (short)0, c1[2], false, false);
        c0[3] = __builtin_amdgcn_wmma_f32_16x16x32_bf16(false, fa3.v, false, fb0.v,
                                                        (short)0, c0[3], false, false);
        c1[3] = __builtin_amdgcn_wmma_f32_16x16x32_bf16(false, fa3.v, false, fb1.v,
                                                        (short)0, c1[3], false, false);
    }

    // write raw conv output + accumulate BN partial sums
    float s0 = 0.0f, q0 = 0.0f, s1 = 0.0f, q1 = 0.0f;
    const int n0 = lane & 15;
    const int n1 = n0 + 16;
    if (base0 + 64 <= N) {
        // full tile: unguarded stores
#pragma unroll
        for (int t = 0; t < 4; ++t) {
#pragma unroll
            for (int r = 0; r < 8; ++r) {
                int pp = base0 + t * 16 + r + half * 8;
                float x0 = c0[t][r];
                float x1 = c1[t][r];
                raw[(long long)pp * 32 + n0] = x0;
                raw[(long long)pp * 32 + n1] = x1;
                s0 += x0; q0 += x0 * x0;
                s1 += x1; q1 += x1 * x1;
            }
        }
    } else {
#pragma unroll
        for (int t = 0; t < 4; ++t) {
#pragma unroll
            for (int r = 0; r < 8; ++r) {
                int pp = base0 + t * 16 + r + half * 8;
                if (pp < N) {
                    float x0 = c0[t][r];
                    float x1 = c1[t][r];
                    raw[(long long)pp * 32 + n0] = x0;
                    raw[(long long)pp * 32 + n1] = x1;
                    s0 += x0; q0 += x0 * x0;
                    s1 += x1; q1 += x1 * x1;
                }
            }
        }
    }
    atomicAdd(&stat[n0],      s0);
    atomicAdd(&stat[32 + n0], q0);
    atomicAdd(&stat[n1],      s1);
    atomicAdd(&stat[32 + n1], q1);
}

// ---------------------------------------------------------------------------
// Fused BN (from accumulated sums) + optional residual + ReLU.
// Writes f32 (residual chain / final output) and/or bf16 (next conv input).
// ---------------------------------------------------------------------------
__global__ void bn_apply(const float* __restrict__ raw,
                         const float* __restrict__ stat,
                         const float* __restrict__ gamma,
                         const float* __restrict__ beta,
                         const float* __restrict__ resid,
                         float*  __restrict__ outf,
                         __bf16* __restrict__ outb,
                         int N) {
    long long i = (long long)blockIdx.x * blockDim.x + threadIdx.x;
    long long total = (long long)N * 32;
    if (i >= total) return;
    int c = (int)(i & 31);
    float invN = 1.0f / (float)N;
    float mu   = stat[c] * invN;
    float var  = stat[32 + c] * invN - mu * mu;
    float sc   = gamma[c] * rsqrtf(var + 1e-5f);
    float sh   = beta[c] - mu * sc;
    float v = fmaf(raw[i], sc, sh);
    if (resid) v += resid[i];
    v = fmaxf(v, 0.0f);
    if (outf) outf[i] = v;
    if (outb) outb[i] = (__bf16)v;
}

// ---------------------------------------------------------------------------
// Launcher
// ---------------------------------------------------------------------------
static inline int cdiv(long long a, long long b) { return (int)((a + b - 1) / b); }

extern "C" void kernel_launch(void* const* d_in, const int* in_sizes, int n_in,
                              void* d_out, int out_size, void* d_ws, size_t ws_size,
                              hipStream_t stream) {
    const float* vf   = (const float*)d_in[0];
    const float* W1   = (const float*)d_in[1];
    const float* W2   = (const float*)d_in[2];
    const float* Wd   = (const float*)d_in[3];
    const float* Wr1a = (const float*)d_in[4];
    const float* Wr1b = (const float*)d_in[5];
    const float* Wr2a = (const float*)d_in[6];
    const float* Wr2b = (const float*)d_in[7];
    const float* gam  = (const float*)d_in[8];
    const float* bet  = (const float*)d_in[9];
    const int* nbr0   = (const int*)d_in[10];
    const int* dn1    = (const int*)d_in[11];
    const int* nbr1   = (const int*)d_in[12];

    const int N0 = in_sizes[0] / 4;    // voxel_features is [N0, 4]
    const int N1 = in_sizes[11] / 8;   // down1 is [8, N1]

    // workspace carve-up (256B aligned)
    char* ws = (char*)d_ws;
    size_t off = 0;
    float* stats = (float*)(ws + off);                 off += ((size_t)7 * 64 * 4 + 255) & ~(size_t)255;
    __bf16* wfrag = (__bf16*)(ws + off);               off += ((size_t)143 * 1024 * 2 + 255) & ~(size_t)255;
    float* raw  = (float*)(ws + off);                  off += (size_t)N0 * 32 * 4;
    __bf16* bfPad = (__bf16*)(ws + off);               off += ((size_t)(N0 + 1) * 32 * 2 + 255) & ~(size_t)255;
    float* x1f  = (float*)(ws + off);                  off += (size_t)N1 * 32 * 4;
    (void)ws_size; (void)n_in; (void)out_size;

    __bf16* bfA = bfPad + 32;                 // features start after pad row

    __bf16* fr_stem2 = wfrag;                 // 27 taps
    __bf16* fr_down  = wfrag + 27  * 1024;    //  8 taps
    __bf16* fr_r1a   = wfrag + 35  * 1024;
    __bf16* fr_r1b   = wfrag + 62  * 1024;
    __bf16* fr_r2a   = wfrag + 89  * 1024;
    __bf16* fr_r2b   = wfrag + 116 * 1024;

    // clear BN stats + the zero-pad feature row (16 f32 == 32 bf16)
    zero_f32<<<cdiv(7 * 64, 256), 256, 0, stream>>>(stats, 7 * 64);
    zero_f32<<<1, 32, 0, stream>>>((float*)bfPad, 16);

    // pre-swizzle weights into WMMA B-fragment layout (bf16)
    prep_w<<<cdiv(27 * 1024, 256), 256, 0, stream>>>(W2,   fr_stem2, 27);
    prep_w<<<cdiv( 8 * 1024, 256), 256, 0, stream>>>(Wd,   fr_down,   8);
    prep_w<<<cdiv(27 * 1024, 256), 256, 0, stream>>>(Wr1a, fr_r1a,   27);
    prep_w<<<cdiv(27 * 1024, 256), 256, 0, stream>>>(Wr1b, fr_r1b,   27);
    prep_w<<<cdiv(27 * 1024, 256), 256, 0, stream>>>(Wr2a, fr_r2a,   27);
    prep_w<<<cdiv(27 * 1024, 256), 256, 0, stream>>>(Wr2b, fr_r2b,   27);

    const int eb0 = cdiv((long long)N0 * 32, 256);   // elementwise grids
    const int eb1 = cdiv((long long)N1 * 32, 256);
    const int cb0 = cdiv(N0, 512);                   // conv grids (8 waves x 64 pts)
    const int cb1 = cdiv(N1, 512);

    // layer 0: stem1 (VALU) + BN/ReLU -> bfA
    stem1_conv<<<eb0, 256, 0, stream>>>(vf, nbr0, W1, raw, stats + 0, N0);
    bn_apply<<<eb0, 256, 0, stream>>>(raw, stats + 0, gam + 0, bet + 0,
                                      nullptr, nullptr, bfA, N0);

    // layer 1: stem2 (WMMA, 27 taps) + BN/ReLU -> bfA
    wmma_conv<<<cb0, 256, 27 * 2048, stream>>>(bfPad, nbr0, (const uint4*)fr_stem2,
                                               raw, stats + 64, N0, 27);
    bn_apply<<<eb0, 256, 0, stream>>>(raw, stats + 64, gam + 32, bet + 32,
                                      nullptr, nullptr, bfA, N0);

    // layer 2: downsample (WMMA, 8 taps, N0 -> N1) + BN/ReLU -> x1f, bfA
    wmma_conv<<<cb1, 256, 8 * 2048, stream>>>(bfPad, dn1, (const uint4*)fr_down,
                                              raw, stats + 128, N1, 8);
    bn_apply<<<eb1, 256, 0, stream>>>(raw, stats + 128, gam + 64, bet + 64,
                                      nullptr, x1f, bfA, N1);

    // layer 3: r1a + BN/ReLU -> bfA
    wmma_conv<<<cb1, 256, 27 * 2048, stream>>>(bfPad, nbr1, (const uint4*)fr_r1a,
                                               raw, stats + 192, N1, 27);
    bn_apply<<<eb1, 256, 0, stream>>>(raw, stats + 192, gam + 96, bet + 96,
                                      nullptr, nullptr, bfA, N1);

    // layer 4: r1b + BN + residual(x1) + ReLU -> x1f, bfA
    wmma_conv<<<cb1, 256, 27 * 2048, stream>>>(bfPad, nbr1, (const uint4*)fr_r1b,
                                               raw, stats + 256, N1, 27);
    bn_apply<<<eb1, 256, 0, stream>>>(raw, stats + 256, gam + 128, bet + 128,
                                      x1f, x1f, bfA, N1);

    // layer 5: r2a + BN/ReLU -> bfA
    wmma_conv<<<cb1, 256, 27 * 2048, stream>>>(bfPad, nbr1, (const uint4*)fr_r2a,
                                               raw, stats + 320, N1, 27);
    bn_apply<<<eb1, 256, 0, stream>>>(raw, stats + 320, gam + 160, bet + 160,
                                      nullptr, nullptr, bfA, N1);

    // layer 6: r2b + BN + residual(x1) + ReLU -> d_out (f32)
    wmma_conv<<<cb1, 256, 27 * 2048, stream>>>(bfPad, nbr1, (const uint4*)fr_r2b,
                                               raw, stats + 384, N1, 27);
    bn_apply<<<eb1, 256, 0, stream>>>(raw, stats + 384, gam + 192, bet + 192,
                                      x1f, (float*)d_out, nullptr, N1);
}
